// TransformerDecoderLayer_14070312861935
// MI455X (gfx1250) — matrix operand order repaired and checked by reading, not compiled, not run on hardware
//
// CDNA5 / gfx1250 implementation of the fused TransformerDecoderLayer scoring head.
//
// Pipeline (all on `stream`, all scratch in d_ws):
//  k_prep    : sigmoid(score_embed) table; split W1, W2 into bf16 hi/lo.
//  k_project : Q = tgt@W1^T+b1, K = mem@W1^T+b1 via bf16x2 WMMA; store bf16 hi/lo.
//  k_score   : per (q,k) pair, 192x192x512 GEMM in registers (144 16x16 WMMA tiles),
//              * sigmoid(embed), dual-axis max-pool via LDS atomic max -> pooled[8192,192],
//              per-block BN1 partial sums (deterministic tree reduce).
//              LDS staging uses GLOBAL_LOAD_ASYNC_TO_LDS_B128 (ASYNCcnt) when available.
//  k_bn1red  : global BN1 scalar stats.
//  k_gram    : G = X^T X (192x192) and column sums of pooled (for analytic BN2 stats).
//  k_coeffs  : fold BN1+b2+BN2 into per-feature affine A_j, C_j.
//  k_mlp     : z = x@W2^T via bf16x2 WMMA, fused affine+ReLU+dot(W3)+pair-sum -> u[4096].
//  k_ustats  : BN3 scalar stats (deterministic).
//  k_final   : BN3 -> out[64,64].
#include <hip/hip_runtime.h>

#define S_    192
#define D_    512
#define DFF_  2048
#define NPAIR 4096
#define NROWS 8192
#define EPS_  1e-5f

// gfx1250 async direct-to-LDS copy path (bypasses VGPRs, tracked by ASYNCcnt).
#if __has_builtin(__builtin_amdgcn_global_load_async_to_lds_b128) && \
    __has_builtin(__builtin_amdgcn_s_wait_asynccnt)
#define USE_ASYNC_LDS 1
#else
#define USE_ASYNC_LDS 0
#endif

typedef __bf16 bf16;
typedef __attribute__((ext_vector_type(16))) __bf16 v16bf;
typedef __attribute__((ext_vector_type(8)))  __bf16 v8bf;
typedef __attribute__((ext_vector_type(8)))  float  v8f;
typedef __attribute__((ext_vector_type(4)))  int    v4i;
// Address-space-qualified pointers for the async builtin: diag shows it wants
// addrspace(1) ("__device__") pointer-to-int4 for the source; LDS side is AS(3).
typedef __attribute__((address_space(1))) v4i* gbl_v4i_p;
typedef __attribute__((address_space(3))) v4i* lds_v4i_p;

static __device__ __forceinline__ v8f zero8() {
  v8f z = {0.f,0.f,0.f,0.f,0.f,0.f,0.f,0.f};
  return z;
}

// fp32 -> bf16 hi + bf16 lo residual (bf16x2 split, ~16 mantissa bits total)
static __device__ __forceinline__ void bf16split(float x, bf16& h, bf16& l) {
  h = (bf16)x;
  l = (bf16)(x - (float)h);
}

// A-fragment 16x32 bf16 from row-major [M][K] storage, row stride RS elements.
// ISA 7.12.2: lane l -> M = l&15, half h = l>>4; elems 0..7 hold K in [8h,8h+8),
// elems 8..15 hold K in [16+8h, 16+8h+8). Two aligned 16B LDS/global loads.
static __device__ __forceinline__ v16bf loadA(const bf16* base, int RS, int lane) {
  int m = lane & 15, h = lane >> 4;
  const bf16* r = base + m * RS;
  v8bf lo = *reinterpret_cast<const v8bf*>(r + 8 * h);
  v8bf hi = *reinterpret_cast<const v8bf*>(r + 16 + 8 * h);
  v16bf a;
#pragma unroll
  for (int i = 0; i < 8; ++i) { a[i] = lo[i]; a[i + 8] = hi[i]; }
  return a;
}

// B-fragment 32x16 bf16 (KxN); source holds B^T row-major (row = N, contiguous K).
// lane l -> N = l&15, half h = l>>4; elems 0..7 hold K in [8h,8h+8), elems 8..15 hold K in [16+8h,16+8h+8).
static __device__ __forceinline__ v16bf loadB(const bf16* base, int RS, int lane) {
  int n = lane & 15, h = lane >> 4;
  const bf16* r = base + n * RS;
  v8bf lo = *reinterpret_cast<const v8bf*>(r + 8 * h);
  v8bf hi = *reinterpret_cast<const v8bf*>(r + 16 + 8 * h);
  v16bf b;
#pragma unroll
  for (int i = 0; i < 8; ++i) { b[i] = lo[i]; b[i + 8] = hi[i]; }
  return b;
}

// bf16x2 emulated-fp32 MAC: (ah+al)*(bh+bl) ~= al*bh + ah*bl + ah*bh (drop lo*lo)
static __device__ __forceinline__ v8f wmma3(v16bf ah, v16bf al, v16bf bh, v16bf bl, v8f c) {
  c = __builtin_amdgcn_wmma_f32_16x16x32_bf16(false, al, false, bh, (short)0, c, false, false);
  c = __builtin_amdgcn_wmma_f32_16x16x32_bf16(false, ah, false, bl, (short)0, c, false, false);
  c = __builtin_amdgcn_wmma_f32_16x16x32_bf16(false, ah, false, bh, (short)0, c, false, false);
  return c;
}

// Order-invariant float atomic max (int trick; init target to -inf).
static __device__ __forceinline__ void atomicMaxF(float* a, float v) {
  if (v >= 0.f) atomicMax(reinterpret_cast<int*>(a), __float_as_int(v));
  else          atomicMin(reinterpret_cast<unsigned int*>(a), __float_as_uint(v));
}

// ---------------------------------------------------------------- k_prep
__global__ void k_prep(const float* se, const float* W1, const float* W2,
                       float* sig, bf16* W1h, bf16* W1l, bf16* W2h, bf16* W2l) {
  int i = blockIdx.x * 256 + threadIdx.x;            // grid covers DFF_*S_ exactly
  if (i < S_ * S_)   sig[i] = 1.f / (1.f + __expf(-se[i]));
  if (i < D_ * D_)   bf16split(W1[i], W1h[i], W1l[i]);
  if (i < DFF_ * S_) bf16split(W2[i], W2h[i], W2l[i]);
}

// ---------------------------------------------------------------- k_project
// out[m,n] = sum_d X[m,d] * W1[n,d] + b1[n]; 128x64 tile per WG, 8 waves, K-chunk 32.
__global__ void k_project(const float* tgt, const float* mem, const float* b1,
                          const bf16* W1h, const bf16* W1l,
                          bf16* Qh, bf16* Ql, bf16* Kh, bf16* Kl) {
  extern __shared__ char smem[];
  const int RS = 40;                                  // padded row stride (16B multiple)
  bf16* xh = (bf16*)smem;                             // [128][RS]
  bf16* xl = xh + 128 * RS;
  int tid = threadIdx.x, lane = tid & 31, w = tid >> 5;
  int nblk = blockIdx.x & 7, mblk = blockIdx.x >> 3;
  int m0 = mblk * 128;
  const float* X; bf16 *Oh, *Ol;
  if (m0 < 12288) { X = tgt; Oh = Qh; Ol = Ql; }
  else            { X = mem; Oh = Kh; Ol = Kl; m0 -= 12288; }
  int n0 = nblk * 64;

  v8f acc[4];
#pragma unroll
  for (int i = 0; i < 4; ++i) acc[i] = zero8();

  for (int d0 = 0; d0 < D_; d0 += 32) {
    { // stage 128x32 fp32 chunk as bf16 hi/lo
      int row = tid >> 1, c = (tid & 1) * 16;
      const float* src = X + (size_t)(m0 + row) * D_ + d0 + c;
#pragma unroll
      for (int i = 0; i < 16; ++i)
        bf16split(src[i], xh[row * RS + c + i], xl[row * RS + c + i]);
    }
    __syncthreads();
    v16bf ah = loadA(xh + (w * 16) * RS, RS, lane);
    v16bf al = loadA(xl + (w * 16) * RS, RS, lane);
#pragma unroll
    for (int nt = 0; nt < 4; ++nt) {
      v16bf bh = loadB(W1h + (size_t)(n0 + nt * 16) * D_ + d0, D_, lane);
      v16bf bl = loadB(W1l + (size_t)(n0 + nt * 16) * D_ + d0, D_, lane);
      acc[nt] = wmma3(ah, al, bh, bl, acc[nt]);
    }
    __syncthreads();
  }
  int h = lane >> 4, nn = lane & 15;
#pragma unroll
  for (int nt = 0; nt < 4; ++nt) {
    int n = n0 + nt * 16 + nn;
    float bias = b1[n];
#pragma unroll
    for (int r = 0; r < 8; ++r) {                     // C layout: M = r + 8h, N = lane&15
      int row = m0 + w * 16 + r + 8 * h;
      bf16 hh, ll; bf16split(acc[nt][r] + bias, hh, ll);
      Oh[(size_t)row * D_ + n] = hh;
      Ol[(size_t)row * D_ + n] = ll;
    }
  }
}

// ---------------------------------------------------------------- k_score
// One WG per (q,k) pair: Score = Key_k @ Query_q^T (192x192, K=512), 144 tiles
// held in VGPRs (18 per wave), fused sigmoid-scale + dual-axis max-pool.
__global__ void k_score(const bf16* Qh, const bf16* Ql, const bf16* Kh, const bf16* Kl,
                        const float* sig, float* pooled, float* bn1p) {
  extern __shared__ char smem[];
  const int RS = 72;                                  // padded (16B multiple)
  const int ASZ = S_ * RS;                            // 13824 bf16
  bf16* sQh = (bf16*)smem;
  bf16* sQl = sQh + ASZ;
  bf16* sKh = sQl + ASZ;
  bf16* sKl = sKh + ASZ;
  float* rowmax = (float*)(smem + 4 * (size_t)ASZ * sizeof(bf16)); // [192] max over t
  float* colmax = rowmax + S_;                                     // [192] max over s

  int tid = threadIdx.x, lane = tid & 31, w = tid >> 5;
  int p = blockIdx.x, q = p >> 6, k = p & 63;
  const bf16* gp[4] = { Qh + (size_t)q * S_ * D_, Ql + (size_t)q * S_ * D_,
                        Kh + (size_t)k * S_ * D_, Kl + (size_t)k * S_ * D_ };
  bf16* lp[4] = { sQh, sQl, sKh, sKl };

  for (int i = tid; i < 2 * S_; i += 256) rowmax[i] = -INFINITY;

  int s0a[18], t0a[18];
#pragma unroll
  for (int ti = 0; ti < 18; ++ti) {
    int tix = ti * 8 + w;
    s0a[ti] = (tix / 12) * 16;
    t0a[ti] = (tix % 12) * 16;
  }
  v8f acc[18];
#pragma unroll
  for (int i = 0; i < 18; ++i) acc[i] = zero8();

  for (int d0 = 0; d0 < D_; d0 += 64) {
    __syncthreads();
    for (int i = tid; i < 6144; i += 256) {           // stage 4 x [192][64] bf16 chunks
      int arr = i / 1536, rem = i - arr * 1536, row = rem >> 3, c = (rem & 7) * 8;
#if USE_ASYNC_LDS
      // Direct global->LDS async copy (no VGPR round-trip), 16B per lane.
      __builtin_amdgcn_global_load_async_to_lds_b128(
          (gbl_v4i_p)(gp[arr] + (size_t)row * D_ + d0 + c),
          (lds_v4i_p)(lp[arr] + row * RS + c),
          0, 0);
#else
      *reinterpret_cast<v8bf*>(lp[arr] + row * RS + c) =
          *reinterpret_cast<const v8bf*>(gp[arr] + (size_t)row * D_ + d0 + c);
#endif
    }
#if USE_ASYNC_LDS
    __builtin_amdgcn_s_wait_asynccnt(0);
#endif
    __syncthreads();
#pragma unroll
    for (int ti = 0; ti < 18; ++ti) {
#pragma unroll
      for (int kc = 0; kc < 64; kc += 32) {
        v16bf ah = loadA(sKh + s0a[ti] * RS + kc, RS, lane);   // A = Key rows (s)
        v16bf al = loadA(sKl + s0a[ti] * RS + kc, RS, lane);
        v16bf bh = loadB(sQh + t0a[ti] * RS + kc, RS, lane);   // B = Query^T (t cols)
        v16bf bl = loadB(sQl + t0a[ti] * RS + kc, RS, lane);
        acc[ti] = wmma3(ah, al, bh, bl, acc[ti]);
      }
    }
  }
  int h = lane >> 4, nn = lane & 15;
#pragma unroll
  for (int ti = 0; ti < 18; ++ti) {
#pragma unroll
    for (int r = 0; r < 8; ++r) {
      int s = s0a[ti] + r + 8 * h, t = t0a[ti] + nn;
      float v = acc[ti][r] * sig[s * S_ + t];         // scale BEFORE max (as reference)
      atomicMaxF(&rowmax[s], v);                      // max over t
      atomicMaxF(&colmax[t], v);                      // max over s
    }
  }
  __syncthreads();
  float* red  = (float*)smem;                         // reuse chunk area
  float* red2 = red + 256;
  float sm = 0.f, sq = 0.f;
  if (tid < S_) {
    float cm = colmax[tid], rm = rowmax[tid];
    pooled[((size_t)(2 * p))     * S_ + tid] = cm;    // row 2p   = max over s
    pooled[((size_t)(2 * p + 1)) * S_ + tid] = rm;    // row 2p+1 = max over t
    sm = cm + rm; sq = cm * cm + rm * rm;
  }
  red[tid] = sm; red2[tid] = sq;
  __syncthreads();
  for (int st = 128; st > 0; st >>= 1) {
    if (tid < st) { red[tid] += red[tid + st]; red2[tid] += red2[tid + st]; }
    __syncthreads();
  }
  if (tid == 0) { bn1p[p] = red[0]; bn1p[NPAIR + p] = red2[0]; }
}

// ---------------------------------------------------------------- k_bn1red
__global__ void k_bn1red(const float* bn1p, float* stats) {
  __shared__ float r1[256], r2[256];
  int tid = threadIdx.x;
  float s = 0.f, ss = 0.f;
  for (int i = tid; i < NPAIR; i += 256) { s += bn1p[i]; ss += bn1p[NPAIR + i]; }
  r1[tid] = s; r2[tid] = ss;
  __syncthreads();
  for (int st = 128; st > 0; st >>= 1) {
    if (tid < st) { r1[tid] += r1[tid + st]; r2[tid] += r2[tid + st]; }
    __syncthreads();
  }
  if (tid == 0) { stats[0] = r1[0]; stats[1] = r2[0]; }
}

// ---------------------------------------------------------------- k_gram
// G[a,b] = sum_r x[r,a]*x[r,b]; colsum[a] = sum_r x[r,a]. One thread per G entry.
__global__ void k_gram(const float* pooled, float* G, float* colsum) {
  int g = blockIdx.x * 256 + threadIdx.x;             // < 36864
  int a = g / S_, b = g - a * S_;
  float acc = 0.f, cs = 0.f;
  for (int r = 0; r < NROWS; ++r) {
    float xa = pooled[(size_t)r * S_ + a];
    float xb = pooled[(size_t)r * S_ + b];
    acc = fmaf(xa, xb, acc);
    cs += xa;
  }
  G[g] = acc;
  if (b == 0) colsum[a] = cs;
}

// ---------------------------------------------------------------- k_coeffs
// Fold BN1 (global affine alpha,beta) + b2 + BN2 into y -> A_j*z + C_j on raw z=x@W2^T.
__global__ void k_coeffs(const float* W2, const float* b2, const float* g2, const float* be2,
                         const float* colsum, const float* G, const float* stats,
                         const float* g1, const float* be1, float* Ac, float* Cc) {
  int j = blockIdx.x * 256 + threadIdx.x;
  const float N1 = (float)NROWS * (float)S_;
  float mu1 = stats[0] / N1;
  float var1 = stats[1] / N1 - mu1 * mu1;
  float alpha = g1[0] * rsqrtf(var1 + EPS_);
  float beta  = be1[0] - alpha * mu1;
  const float* wr = W2 + (size_t)j * S_;
  float s = 0.f, m = 0.f;
  for (int a = 0; a < S_; ++a) { float wa = wr[a]; s += wa; m = fmaf(wa, colsum[a], m); }
  float qv = 0.f;
  for (int a = 0; a < S_; ++a) {
    float wa = wr[a], t = 0.f;
    for (int b = 0; b < S_; ++b) t = fmaf(G[a * S_ + b], wr[b], t);
    qv = fmaf(wa, t, qv);
  }
  const float Nf = (float)NROWS;
  float c0 = beta * s + b2[j];                        // y = alpha*z + c0
  float mu = (alpha * m) / Nf + c0;                   // column mean of y
  float Sy2 = alpha * alpha * qv + 2.f * alpha * c0 * m + Nf * c0 * c0;
  float var = Sy2 / Nf - mu * mu;                     // biased column variance of y
  float aj = g2[j] * rsqrtf(var + EPS_);
  Ac[j] = aj * alpha;
  Cc[j] = aj * (c0 - mu) + be2[j];
}

// ---------------------------------------------------------------- k_mlp
// Fused z=x@W2^T -> affine -> ReLU -> *W3 -> row sum -> pair sum. 128 rows/WG.
__global__ void k_mlp(const float* pooled, const bf16* W2h, const bf16* W2l,
                      const float* Ac, const float* Cc, const float* W3, const float* b3,
                      float* u) {
  extern __shared__ char smem[];
  const int RS = 200;                                 // padded row stride
  bf16* xh = (bf16*)smem;                             // [128][RS]
  bf16* xl = xh + 128 * RS;
  float* rowacc = (float*)(xl + 128 * RS);            // [128]
  int tid = threadIdx.x, lane = tid & 31, w = tid >> 5;
  int r0 = blockIdx.x * 128;
  for (int i = tid; i < 128 * 48; i += 256) {         // stage full K=192 as bf16 hi/lo
    int row = i / 48, c = (i - row * 48) * 4;
    const float* src = pooled + (size_t)(r0 + row) * S_ + c;
#pragma unroll
    for (int t = 0; t < 4; ++t)
      bf16split(src[t], xh[row * RS + c + t], xl[row * RS + c + t]);
  }
  if (tid < 128) rowacc[tid] = 0.f;
  __syncthreads();

  float part[8] = {};
  int h = lane >> 4, nn = lane & 15;
  for (int jb = 0; jb < DFF_; jb += 64) {
    v8f acc[4];
#pragma unroll
    for (int i = 0; i < 4; ++i) acc[i] = zero8();
#pragma unroll
    for (int kk = 0; kk < S_; kk += 32) {
      v16bf ah = loadA(xh + (w * 16) * RS + kk, RS, lane);
      v16bf al = loadA(xl + (w * 16) * RS + kk, RS, lane);
#pragma unroll
      for (int nt = 0; nt < 4; ++nt) {
        int j0 = jb + nt * 16;
        v16bf bh = loadB(W2h + (size_t)j0 * S_ + kk, S_, lane);
        v16bf bl = loadB(W2l + (size_t)j0 * S_ + kk, S_, lane);
        acc[nt] = wmma3(ah, al, bh, bl, acc[nt]);
      }
    }
#pragma unroll
    for (int nt = 0; nt < 4; ++nt) {
      int j = jb + nt * 16 + nn;
      float aj = Ac[j], cj = Cc[j], w3 = W3[j];
#pragma unroll
      for (int r = 0; r < 8; ++r) {
        float v = fmaf(acc[nt][r], aj, cj);
        v = fmaxf(v, 0.f);
        part[r] = fmaf(w3, v, part[r]);
      }
    }
  }
  // deterministic cross-lane sum over the 16 lanes sharing a row (j dimension)
#pragma unroll
  for (int r = 0; r < 8; ++r) {
    float v = part[r];
    v += __shfl_xor(v, 1, 32);
    v += __shfl_xor(v, 2, 32);
    v += __shfl_xor(v, 4, 32);
    v += __shfl_xor(v, 8, 32);
    if (nn == 0) rowacc[w * 16 + r + 8 * h] = v;
  }
  __syncthreads();
  if (tid < 64) {
    float uu = rowacc[2 * tid] + rowacc[2 * tid + 1] + 2.f * b3[0];
    u[blockIdx.x * 64 + tid] = uu;
  }
}

// ---------------------------------------------------------------- k_ustats
__global__ void k_ustats(const float* u, float* stats) {
  __shared__ float r1[256], r2[256];
  int tid = threadIdx.x;
  float s = 0.f, ss = 0.f;
  for (int i = tid; i < NPAIR; i += 256) { float v = u[i]; s += v; ss += v * v; }
  r1[tid] = s; r2[tid] = ss;
  __syncthreads();
  for (int st = 128; st > 0; st >>= 1) {
    if (tid < st) { r1[tid] += r1[tid + st]; r2[tid] += r2[tid + st]; }
    __syncthreads();
  }
  if (tid == 0) { stats[2] = r1[0]; stats[3] = r2[0]; }
}

// ---------------------------------------------------------------- k_final
__global__ void k_final(const float* u, const float* stats, const float* g3,
                        const float* be3, float* out) {
  int p = blockIdx.x * 256 + threadIdx.x;
  float mu = stats[2] / (float)NPAIR;
  float var = stats[3] / (float)NPAIR - mu * mu;
  float sc = g3[0] * rsqrtf(var + EPS_);
  out[p] = sc * (u[p] - mu) + be3[0];
}

// ---------------------------------------------------------------- host
extern "C" void kernel_launch(void* const* d_in, const int* in_sizes, int n_in,
                              void* d_out, int out_size, void* d_ws, size_t ws_size,
                              hipStream_t stream) {
  (void)in_sizes; (void)n_in; (void)out_size; (void)ws_size;
  const float* tgt = (const float*)d_in[0];
  const float* mem = (const float*)d_in[1];
  const float* W1  = (const float*)d_in[2];
  const float* b1  = (const float*)d_in[3];
  const float* se  = (const float*)d_in[4];
  const float* g1  = (const float*)d_in[5];
  const float* be1 = (const float*)d_in[6];
  const float* W2  = (const float*)d_in[7];
  const float* b2  = (const float*)d_in[8];
  const float* g2  = (const float*)d_in[9];
  const float* be2 = (const float*)d_in[10];
  const float* W3  = (const float*)d_in[11];
  const float* b3  = (const float*)d_in[12];
  const float* g3  = (const float*)d_in[13];
  const float* be3 = (const float*)d_in[14];
  float* out = (float*)d_out;

  char* ws = (char*)d_ws;
  size_t off = 0;
  auto take = [&](size_t bytes) -> char* {
    char* p = ws + off;
    off += (bytes + 255) & ~(size_t)255;
    return p;
  };
  float* sig    = (float*)take((size_t)S_ * S_ * 4);
  bf16* W1h     = (bf16*)take((size_t)D_ * D_ * 2);
  bf16* W1l     = (bf16*)take((size_t)D_ * D_ * 2);
  bf16* W2h     = (bf16*)take((size_t)DFF_ * S_ * 2);
  bf16* W2l     = (bf16*)take((size_t)DFF_ * S_ * 2);
  bf16* Qh      = (bf16*)take((size_t)64 * S_ * D_ * 2);
  bf16* Ql      = (bf16*)take((size_t)64 * S_ * D_ * 2);
  bf16* Kh      = (bf16*)take((size_t)64 * S_ * D_ * 2);
  bf16* Kl      = (bf16*)take((size_t)64 * S_ * D_ * 2);
  float* pooled = (float*)take((size_t)NROWS * S_ * 4);
  float* bn1p   = (float*)take((size_t)2 * NPAIR * 4);
  float* colsum = (float*)take((size_t)S_ * 4);
  float* G      = (float*)take((size_t)S_ * S_ * 4);
  float* Ac     = (float*)take((size_t)DFF_ * 4);
  float* Cc     = (float*)take((size_t)DFF_ * 4);
  float* u      = (float*)take((size_t)NPAIR * 4);
  float* stats  = (float*)take(8 * 4);

  size_t lds1 = (size_t)2 * 128 * 40 * sizeof(bf16);                         // 20.0 KB
  size_t lds2 = (size_t)4 * S_ * 72 * sizeof(bf16) + 2 * S_ * sizeof(float); // 109.5 KB
  size_t lds5 = (size_t)2 * 128 * 200 * sizeof(bf16) + 128 * sizeof(float);  // 100.5 KB

  k_prep   <<<dim3((DFF_ * S_) / 256), dim3(256), 0,    stream>>>(se, W1, W2, sig, W1h, W1l, W2h, W2l);
  k_project<<<dim3(1536),              dim3(256), lds1, stream>>>(tgt, mem, b1, W1h, W1l, Qh, Ql, Kh, Kl);
  k_score  <<<dim3(NPAIR),             dim3(256), lds2, stream>>>(Qh, Ql, Kh, Kl, sig, pooled, bn1p);
  k_bn1red <<<dim3(1),                 dim3(256), 0,    stream>>>(bn1p, stats);
  k_gram   <<<dim3((S_ * S_) / 256),   dim3(256), 0,    stream>>>(pooled, G, colsum);
  k_coeffs <<<dim3(DFF_ / 256),        dim3(256), 0,    stream>>>(W2, b2, g2, be2, colsum, G, stats, g1, be1, Ac, Cc);
  k_mlp    <<<dim3(NROWS / 128),       dim3(256), lds5, stream>>>(pooled, W2h, W2l, Ac, Cc, W3, b3, u);
  k_ustats <<<dim3(1),                 dim3(256), 0,    stream>>>(u, stats);
  k_final  <<<dim3(NPAIR / 256),       dim3(256), 0,    stream>>>(u, stats, g3, be3, out);
}
